// GemmaAttention_7206955123112
// MI455X (gfx1250) — matrix operand order repaired and checked by reading, not compiled, not run on hardware
//
#include <hip/hip_runtime.h>
#include <hip/hip_bf16.h>

#ifndef __has_builtin
#define __has_builtin(x) 0
#endif

#if defined(__HIP_DEVICE_COMPILE__) && __has_builtin(__builtin_amdgcn_tensor_load_to_lds) && \
    __has_builtin(__builtin_amdgcn_s_wait_tensorcnt)
#define USE_TDM 1
#else
#define USE_TDM 0
#endif

// ---------------- problem constants (from reference) ----------------
constexpr int L_   = 4096;
constexpr int D_   = 2048;
constexpr int H_   = 8;
constexpr int KVH_ = 2;
constexpr int HD_  = 256;
constexpr int WIN_ = 512;
constexpr int HHD_ = H_ * HD_;     // 2048
constexpr int KVHD_ = KVH_ * HD_;  // 512

typedef __attribute__((ext_vector_type(16))) __bf16 v16bf;
typedef __attribute__((ext_vector_type(8)))  __bf16 v8bf;
typedef __attribute__((ext_vector_type(8)))  float  v8f;
typedef __attribute__((ext_vector_type(4)))  unsigned v4u;
typedef __attribute__((ext_vector_type(8)))  int      v8i_;
typedef __attribute__((ext_vector_type(4)))  int      v4i_;

// ---------------- helpers ----------------
__device__ __host__ inline __bf16 f2bf(float f) {
  unsigned u = __builtin_bit_cast(unsigned, f);
  unsigned r = (u + 0x7FFFu + ((u >> 16) & 1u)) >> 16;   // round-to-nearest-even
  return __builtin_bit_cast(__bf16, (unsigned short)r);
}

__device__ inline v16bf cat8(v8bf lo, v8bf hi) {
  v16bf r;
#pragma unroll
  for (int j = 0; j < 8; ++j) { r[j] = lo[j]; r[j + 8] = hi[j]; }
  return r;
}

// A-fragment (16x32 bf16): lane holds row (lane&15); low lanes K = {0..15},
// high lanes K = {16..31}.  rowp points at the 32-deep K row.
__device__ inline v16bf loadA_frag(const __bf16* rowp, int base) {
  return cat8(*(const v8bf*)(rowp + 2 * base), *(const v8bf*)(rowp + 2 * base + 8));
}
// B-fragment (32x16 bf16) when the 16 K-values for this half-wave are contiguous.
__device__ inline v16bf loadB_contig(const __bf16* p) {
  return cat8(*(const v8bf*)p, *(const v8bf*)(p + 8));
}

__device__ inline v8f wmma_bf16(v16bf a, v16bf b, v8f c) {
  return __builtin_amdgcn_wmma_f32_16x16x32_bf16(
      /*neg_a=*/false, a, /*neg_b=*/false, b,
      /*c_mod=*/(short)0, c, /*reuse_a=*/false, /*reuse_b=*/false);
}

#if USE_TDM
// Issue a TDM 2-D tile load: `rows` rows of 32 bf16 (64B) each, row stride
// `row_stride_elems`, into LDS at byte offset `lds_off`, with hardware padding
// of 4 DWORDs (8 elems) after every 16 DWORDs (32 elems) -> 80B LDS rows.
// D# layout per CDNA5 ISA ch.8 (group0 128b, group1 256b; trailing groups zero
// -> 2D tile, tile_dim2..4 = 0).  6-arg builtin on this toolchain.
__device__ inline void tdm_load_tile32(const __bf16* gsrc, unsigned lds_off,
                                       int rows, int row_stride_elems) {
  unsigned long long ga = (unsigned long long)(const void*)gsrc;
  unsigned K = (unsigned)row_stride_elems;
  unsigned R = (unsigned)rows;

  v4u g0;
  g0[0] = 1u;                                        // count=1 (valid user D#)
  g0[1] = lds_off;                                   // lds_addr [63:32]
  g0[2] = (unsigned)(ga & 0xFFFFFFFFull);            // global_addr lo
  g0[3] = (unsigned)((ga >> 32) & 0x01FFFFFFull)     // global_addr [56:32]
          | (2u << 30);                              // type = 2 ("image")

  v8i_ g1;
  g1[0] = (int)((1u << 16)       // data_size = 1 -> 2 bytes
              | (1u << 20)       // pad_enable
              | (3u << 22)       // pad_interval: code 3 -> every 16 DWORDs
              | (3u << 25));     // pad_amount:   code 3 -> 4 DWORDs
  g1[1] = (int)((K & 0xFFFFu) << 16);                        // tensor_dim0 lo16 @ [63:48]
  g1[2] = (int)((K >> 16) | ((R & 0xFFFFu) << 16));          // dim0 hi16, tensor_dim1 lo16
  g1[3] = (int)((R >> 16) | (32u << 16));                    // dim1 hi16, tile_dim0 = 32
  g1[4] = (int)(R & 0xFFFFu);                                // tile_dim1 = rows, tile_dim2 = 0
  g1[5] = (int)K;                                            // tensor_dim0_stride lo32
  g1[6] = 0;                                                 // stride hi16 = 0, dim1_stride lo = 0
  g1[7] = 0;

  v4i_ z4 = {0, 0, 0, 0};
  v8i_ z8 = {0, 0, 0, 0, 0, 0, 0, 0};
  __builtin_amdgcn_tensor_load_to_lds(g0, g1, z4, z4, z8, 0);
}
__device__ inline unsigned lds_offset(const void* p) {
  return (unsigned)(unsigned long long)p;            // flat LDS addr: low 32b = offset
}
#endif

// ---------------- kernel 1a: fp32 -> bf16 convert (same layout) ----------------
__global__ void cvt_bf16(const float* __restrict__ src, __bf16* __restrict__ dst, int n) {
  int i = blockIdx.x * blockDim.x + threadIdx.x;
  int stride = gridDim.x * blockDim.x;
  for (; i < n; i += stride) dst[i] = f2bf(src[i]);
}

// ---------------- kernel 1b: fp32 (KxN) -> bf16 transposed (NxK) ----------------
__global__ void cvt_bf16_t(const float* __restrict__ src, __bf16* __restrict__ dst,
                           int K, int N) {
  int i = blockIdx.x * blockDim.x + threadIdx.x;
  int stride = gridDim.x * blockDim.x;
  int total = K * N;
  for (; i < total; i += stride) {
    int k = i / N, n = i - k * N;               // coalesced read of src
    dst[(size_t)n * K + k] = f2bf(src[i]);
  }
}

// ---------------- kernel 2: bf16 GEMM, C = A(MxK) * BT(NxK)^T, fp32 out --------
// workgroup: 256 threads = 8 waves; tile 64(M) x 256(N); each wave 32x64.
// Double-buffered TDM staging: wave 0 issues next tile-pair, waits tensorcnt<=2
// for the previous pair (in-order completion), hiding DMA behind WMMA compute.
constexpr int LPAD = 40;   // 32 K elems + 8 pad (80B rows, 16B aligned)
__global__ __launch_bounds__(256) void gemm_bf16_tn(const __bf16* __restrict__ A,
                                                    const __bf16* __restrict__ BT,
                                                    float* __restrict__ C,
                                                    int M, int N, int K) {
  __shared__ __align__(16) __bf16 As[2][64][LPAD];
  __shared__ __align__(16) __bf16 BsT[2][256][LPAD];

  const int tid  = threadIdx.x;
  const int lane = tid & 31;
  const int w    = tid >> 5;
  const int wr   = w >> 2;            // 0..1 -> 32-row slab
  const int wc   = w & 3;             // 0..3 -> 64-col slab
  const int lm   = lane & 15;
  const int hi   = lane >> 4;         // 0 or 1
  const int base = hi ? 8 : 0;        // A-frag K base
  const int koff = hi ? 16 : 0;       // B-frag K base
  const int mbase = blockIdx.y * 64;
  const int nbase = blockIdx.x * 256;

#if !USE_TDM
  const int ar = (tid * 8) >> 5, ac = (tid * 8) & 31;     // A: 8 elems/thread
#endif

  v8f acc[2][4] = {};

#if USE_TDM
  if (w == 0) {                         // prologue: stage K-step 0 into buffer 0
    tdm_load_tile32(&A[(size_t)mbase * K],  lds_offset(&As[0][0][0]),  64,  K);
    tdm_load_tile32(&BT[(size_t)nbase * K], lds_offset(&BsT[0][0][0]), 256, K);
  }
#endif

  for (int k0 = 0; k0 < K; k0 += 32) {
    const int buf = (k0 >> 5) & 1;
    __syncthreads();                    // all waves done with the other buffer
#if USE_TDM
    if (w == 0) {
      if (k0 + 32 < K) {                // pipeline: issue next pair, wait prev pair
        tdm_load_tile32(&A[(size_t)mbase * K + k0 + 32],
                        lds_offset(&As[buf ^ 1][0][0]), 64, K);
        tdm_load_tile32(&BT[(size_t)nbase * K + k0 + 32],
                        lds_offset(&BsT[buf ^ 1][0][0]), 256, K);
        __builtin_amdgcn_s_wait_tensorcnt(2);
      } else {
        __builtin_amdgcn_s_wait_tensorcnt(0);
      }
    }
#else
    {   // stage A tile 64x32 into buf
      const __bf16* ap = &A[(size_t)(mbase + ar) * K + k0 + ac];
      *(v8bf*)&As[buf][ar][ac] = *(const v8bf*)ap;
      if (k0 + 32 < K) __builtin_prefetch(ap + 32, 0, 1);
    }
    {   // stage BT tile 256x32 into buf: one row per thread (4 x 16B)
      const __bf16* bp = &BT[(size_t)(nbase + tid) * K + k0];
#pragma unroll
      for (int j = 0; j < 4; ++j)
        *(v8bf*)&BsT[buf][tid][j * 8] = *(const v8bf*)(bp + j * 8);
      if (k0 + 32 < K) __builtin_prefetch(bp + 32, 0, 1);
    }
#endif
    __syncthreads();                    // current buffer visible to all waves

    v16bf af[2], bfr[4];
#pragma unroll
    for (int mt = 0; mt < 2; ++mt)
      af[mt] = loadA_frag(&As[buf][wr * 32 + mt * 16 + lm][0], base);
#pragma unroll
    for (int nt = 0; nt < 4; ++nt)
      bfr[nt] = loadB_contig(&BsT[buf][wc * 64 + nt * 16 + lm][koff]);
#pragma unroll
    for (int mt = 0; mt < 2; ++mt)
#pragma unroll
      for (int nt = 0; nt < 4; ++nt)
        acc[mt][nt] = wmma_bf16(af[mt], bfr[nt], acc[mt][nt]);
  }

#pragma unroll
  for (int mt = 0; mt < 2; ++mt)
#pragma unroll
    for (int nt = 0; nt < 4; ++nt) {
      int row0 = mbase + wr * 32 + mt * 16 + hi * 8;
      int col  = nbase + wc * 64 + nt * 16 + lm;
#pragma unroll
      for (int r = 0; r < 8; ++r)
        C[(size_t)(row0 + r) * N + col] = acc[mt][nt][r];
    }
}

// ---------------- kernel 3: RMSNorm (+scale) + RoPE, fp32 in, bf16 out --------
// grid.x = token l, grid.y in [0,12): 0..7 q heads, 8..9 k heads, 10..11 v heads.
// V is written TRANSPOSED: vt[(kvh*HD + d) * L + l]  (K-contiguous for PV WMMA).
__global__ __launch_bounds__(256) void norm_rope(const float* __restrict__ qf,
                                                 const float* __restrict__ kf,
                                                 const float* __restrict__ vf,
                                                 const float* __restrict__ q_scale,
                                                 const float* __restrict__ k_scale,
                                                 const int* __restrict__ positions,
                                                 __bf16* __restrict__ qb,
                                                 __bf16* __restrict__ kb,
                                                 __bf16* __restrict__ vt) {
  __shared__ float red[256];
  __shared__ float sh[256];
  const int l = blockIdx.x, y = blockIdx.y, d = threadIdx.x;

  const float* src; float g;
  if (y < H_) {
    src = qf + ((size_t)l * H_ + y) * HD_;
    g = q_scale[d];
  } else if (y < H_ + KVH_) {
    src = kf + ((size_t)l * KVH_ + (y - H_)) * HD_;
    g = k_scale[d];
  } else {
    src = vf + ((size_t)l * KVH_ + (y - H_ - KVH_)) * HD_;
    g = 1.0f;
  }

  float val = src[d];
  red[d] = val * val;
  __syncthreads();
#pragma unroll
  for (int s = 128; s > 0; s >>= 1) {
    if (d < s) red[d] += red[d + s];
    __syncthreads();
  }
  float rs = rsqrtf(red[0] * (1.0f / HD_) + 1e-6f);
  float xn = val * rs * g;

  if (y >= H_ + KVH_) {                 // V path: no rope, transposed store
    int h = y - H_ - KVH_;
    vt[((size_t)h * HD_ + d) * L_ + l] = f2bf(xn);
    return;
  }

  sh[d] = xn;
  __syncthreads();
  const int half = HD_ / 2;             // 128, rope_fraction == 1
  int i = d & (half - 1);
  float x1 = sh[i], x2 = sh[i + half];
  float pos = (float)positions[l];
  float inv_ts = __expf(-(float)i * (9.2103403719761836f / (float)half));
  float sv, cv;
  __sincosf(pos * inv_ts, &sv, &cv);
  float out = (d < half) ? (x1 * cv - x2 * sv) : (x2 * cv + x1 * sv);

  if (y < H_) qb[((size_t)l * H_ + y) * HD_ + d] = f2bf(out);
  else        kb[((size_t)l * KVH_ + (y - H_)) * HD_ + d] = f2bf(out);
}

// ---------------- kernel 4: sliding-window flash attention --------------------
// one wave (32 threads) per (16-query tile, head). grid = (L/16, H)
__global__ __launch_bounds__(32) void attention(const __bf16* __restrict__ qb,
                                                const __bf16* __restrict__ kb,
                                                const __bf16* __restrict__ vt,
                                                __bf16* __restrict__ att) {
  __shared__ __align__(16) __bf16 psh[16][LPAD];   // P transpose buffer (padded)

  const int t    = blockIdx.x;          // query tile (16 rows)
  const int h    = blockIdx.y;          // query head
  const int kvh  = h / (H_ / KVH_);     // GQA: kv head = h/4
  const int lane = threadIdx.x;
  const int lm   = lane & 15;
  const int hi   = lane >> 4;
  const int base = hi ? 8 : 0;
  const int koff = hi ? 16 : 0;

  v8f acc[16] = {};                     // 16 q-rows x 256 hd (fp32)
  float mrow[8], lrow[8];
#pragma unroll
  for (int r = 0; r < 8; ++r) { mrow[r] = -1e30f; lrow[r] = 0.0f; }

  int kc0 = t * 16 - (WIN_ - 1);
  if (kc0 < 0) kc0 = 0;
  kc0 = (kc0 / 32) * 32;
  const int kc1 = t * 16 + 16;          // keys <= last query row

  const __bf16* qrowp = qb + ((size_t)(t * 16 + lm) * H_ + h) * HD_;
  const __bf16* vtp   = vt + (size_t)kvh * HD_ * L_;   // [hd][L], K-contiguous

  for (int kc = kc0; kc < kc1; kc += 32) {
    // ---- S = Q K^T over the 32-key chunk (two 16x16 tiles) ----
    v8f s0 = {}, s1 = {};
    const __bf16* kp0 = kb + ((size_t)(kc + lm) * KVH_ + kvh) * HD_;
    const __bf16* kp1 = kb + ((size_t)(kc + 16 + lm) * KVH_ + kvh) * HD_;
    if (kc + 32 < kc1) {
      __builtin_prefetch(kp0 + (size_t)32 * KVHD_, 0, 1);   // next K chunk
      __builtin_prefetch(kp1 + (size_t)32 * KVHD_, 0, 1);
    }
#pragma unroll
    for (int kd = 0; kd < HD_; kd += 32) {
      v16bf a  = loadA_frag(qrowp + kd, base);
      v16bf b0 = loadB_contig(kp0 + kd + koff);
      v16bf b1 = loadB_contig(kp1 + kd + koff);
      s0 = wmma_bf16(a, b0, s0);
      s1 = wmma_bf16(a, b1, s1);
    }

    // ---- mask + online softmax (row stats per lane for its 8 rows) ----
    float p0[8], p1[8], alpha[8];
#pragma unroll
    for (int r = 0; r < 8; ++r) {
      float v0 = s0[r], v1 = s1[r];
      int qi  = t * 16 + r + hi * 8;
      int k0i = kc + lm;
      int k1i = kc + 16 + lm;
      if (k0i > qi || k0i < qi - (WIN_ - 1)) v0 = -10000.0f;
      if (k1i > qi || k1i < qi - (WIN_ - 1)) v1 = -10000.0f;
      float cm = fmaxf(v0, v1);
#pragma unroll
      for (int off = 8; off >= 1; off >>= 1)     // reduce within 16-lane halves
        cm = fmaxf(cm, __shfl_xor(cm, off, 32));
      float mn = fmaxf(mrow[r], cm);
      alpha[r] = __expf(mrow[r] - mn);
      p0[r] = __expf(v0 - mn);
      p1[r] = __expf(v1 - mn);
      float ps = p0[r] + p1[r];
#pragma unroll
      for (int off = 8; off >= 1; off >>= 1)
        ps += __shfl_xor(ps, off, 32);
      lrow[r] = lrow[r] * alpha[r] + ps;
      mrow[r] = mn;
    }

    // rescale running accumulators
#pragma unroll
    for (int nt = 0; nt < 16; ++nt)
#pragma unroll
      for (int r = 0; r < 8; ++r)
        acc[nt][r] *= alpha[r];

    // ---- P -> LDS (C layout) then reload as A-fragment (transpose) ----
    __syncthreads();
#pragma unroll
    for (int r = 0; r < 8; ++r) {
      psh[r + hi * 8][lm]      = f2bf(p0[r]);
      psh[r + hi * 8][16 + lm] = f2bf(p1[r]);
    }
    __syncthreads();
    v16bf pa = loadA_frag(&psh[lm][0], base);

    // ---- O += P * V : B-fragments are key-contiguous loads from V^T ----
#pragma unroll
    for (int nt = 0; nt < 16; ++nt) {
      v16bf bv = loadB_contig(vtp + (size_t)(nt * 16 + lm) * L_ + kc + koff);
      acc[nt] = wmma_bf16(pa, bv, acc[nt]);
    }
  }

  // ---- normalize and store bf16 for the Wo GEMM ----
#pragma unroll
  for (int nt = 0; nt < 16; ++nt) {
    int col = h * HD_ + nt * 16 + lm;
#pragma unroll
    for (int r = 0; r < 8; ++r) {
      int row = t * 16 + r + hi * 8;
      att[(size_t)row * HHD_ + col] = f2bf(acc[nt][r] / lrow[r]);
    }
  }
}

// ---------------- host launch ----------------
extern "C" void kernel_launch(void* const* d_in, const int* in_sizes, int n_in,
                              void* d_out, int out_size, void* d_ws, size_t ws_size,
                              hipStream_t stream) {
  (void)in_sizes; (void)n_in; (void)out_size; (void)ws_size;
  const float* x         = (const float*)d_in[0];
  const int*   positions = (const int*)  d_in[1];
  const float* Wq        = (const float*)d_in[2];
  const float* Wk        = (const float*)d_in[3];
  const float* Wv        = (const float*)d_in[4];
  const float* Wo        = (const float*)d_in[5];
  const float* q_scale   = (const float*)d_in[6];
  const float* k_scale   = (const float*)d_in[7];
  float* out = (float*)d_out;

  char* ws = (char*)d_ws;
  size_t off = 0;
  auto alloc = [&](size_t bytes) { char* p = ws + off; off += (bytes + 255) & ~(size_t)255; return p; };

  __bf16* xb  = (__bf16*)alloc((size_t)L_ * D_ * 2);        // x, row-major (A)
  __bf16* wqt = (__bf16*)alloc((size_t)HHD_ * D_ * 2);      // Wq^T  [N][K]
  __bf16* wkt = (__bf16*)alloc((size_t)KVHD_ * D_ * 2);     // Wk^T
  __bf16* wvt = (__bf16*)alloc((size_t)KVHD_ * D_ * 2);     // Wv^T
  __bf16* wot = (__bf16*)alloc((size_t)D_ * HHD_ * 2);      // Wo^T
  float*  qf  = (float*) alloc((size_t)L_ * HHD_ * 4);
  float*  kf  = (float*) alloc((size_t)L_ * KVHD_ * 4);
  float*  vf  = (float*) alloc((size_t)L_ * KVHD_ * 4);
  __bf16* qb2 = (__bf16*)alloc((size_t)L_ * HHD_ * 2);
  __bf16* kb2 = (__bf16*)alloc((size_t)L_ * KVHD_ * 2);
  __bf16* vt  = (__bf16*)alloc((size_t)KVH_ * HD_ * L_ * 2); // V^T [kvh][hd][L]
  __bf16* att = (__bf16*)alloc((size_t)L_ * HHD_ * 2);

  // 1) convert activations (row-major) + weights (transposed) to bf16
  cvt_bf16  <<<1024, 256, 0, stream>>>(x, xb, L_ * D_);
  cvt_bf16_t<<<1024, 256, 0, stream>>>(Wq, wqt, D_, HHD_);
  cvt_bf16_t<<<1024, 256, 0, stream>>>(Wk, wkt, D_, KVHD_);
  cvt_bf16_t<<<1024, 256, 0, stream>>>(Wv, wvt, D_, KVHD_);
  cvt_bf16_t<<<1024, 256, 0, stream>>>(Wo, wot, HHD_, D_);

  // 2) QKV projections (WMMA bf16, fp32 accumulate, double-buffered TDM tiles)
  gemm_bf16_tn<<<dim3(HHD_ / 256, L_ / 64), 256, 0, stream>>>(xb, wqt, qf, L_, HHD_, D_);
  gemm_bf16_tn<<<dim3(KVHD_ / 256, L_ / 64), 256, 0, stream>>>(xb, wkt, kf, L_, KVHD_, D_);
  gemm_bf16_tn<<<dim3(KVHD_ / 256, L_ / 64), 256, 0, stream>>>(xb, wvt, vf, L_, KVHD_, D_);

  // 3) RMSNorm (+scale) and RoPE -> bf16 q/k (row-major) and V^T
  norm_rope<<<dim3(L_, H_ + 2 * KVH_), 256, 0, stream>>>(qf, kf, vf, q_scale, k_scale,
                                                         positions, qb2, kb2, vt);

  // 4) sliding-window attention (flash-style, WMMA scores + PV)
  attention<<<dim3(L_ / 16, H_), 32, 0, stream>>>(qb2, kb2, vt, att);

  // 5) output projection -> d_out (fp32)
  gemm_bf16_tn<<<dim3(D_ / 256, L_ / 64), 256, 0, stream>>>(att, wot, out, L_, D_, HHD_);
}
